// RoPEAttention_5076651344392
// MI455X (gfx1250) — matrix operand order repaired and checked by reading, not compiled, not run on hardware
//
#include <hip/hip_runtime.h>
#include <hip/hip_bf16.h>
#include <math.h>

// ---------------------------------------------------------------------------
// RoPE multi-head attention for gfx1250 (MI455X).
// bf16 WMMA pipeline + Tensor Data Mover (TDM) LDS staging, double-buffered.
// ---------------------------------------------------------------------------

typedef __attribute__((ext_vector_type(16))) __bf16       v16bf;
typedef __attribute__((ext_vector_type(8)))  __bf16       v8bf;
typedef __attribute__((ext_vector_type(8)))  float        v8f;
typedef __attribute__((ext_vector_type(4)))  unsigned int v4u;
typedef __attribute__((ext_vector_type(8)))  int          v8i;
typedef __attribute__((ext_vector_type(4)))  int          v4i;

constexpr int DM   = 1024;   // d_model
constexpr int SEQ  = 2048;   // sequence length
constexpr int NBAT = 2;      // batch
constexpr int NH   = 16;     // heads
constexpr int HD   = 64;     // head dim
constexpr int MTOT = NBAT * SEQ;   // 4096 projection rows

__device__ __forceinline__ __bf16 f2bf(float f) {
  // round-to-nearest-even fp32 -> bf16
  unsigned u = __builtin_bit_cast(unsigned, f);
  unsigned r = (u + 0x7FFFu + ((u >> 16) & 1u)) >> 16;
  unsigned short h = (unsigned short)r;
  return __builtin_bit_cast(__bf16, h);
}

// ---------------------------------------------------------------------------
// Tensor Data Mover: async 2D tile load Global -> LDS (bf16 elements).
// D# built per CDNA5 ISA 8.3/8.4: group0 {count=1, lds_addr, global_addr,
// type=2}; group1 {data_size=1(2B), tensor dims, tile dims, dim0 stride}.
// tile == tensor window here (no OOB clipping needed). 6-arg builtin form
// (clang-23 / therock): groups 2/3 + trailing int32x8 zero-filled for 2D.
// ---------------------------------------------------------------------------
__device__ __forceinline__ void tdm_load_2d(void* lds_ptr, const void* gptr,
                                            unsigned td0,   // contiguous elems
                                            unsigned td1,   // rows
                                            unsigned long long stride) // row stride (elems)
{
  unsigned lds_off =
      (unsigned)(unsigned long long)(__attribute__((address_space(3))) void*)lds_ptr;
  unsigned long long ga = (unsigned long long)gptr;

  v4u g0;
  g0[0] = 1u;                                           // count=1, user mode
  g0[1] = lds_off;                                      // lds_addr (bytes)
  g0[2] = (unsigned)ga;                                 // global_addr[31:0]
  g0[3] = (unsigned)((ga >> 32) & 0x01FFFFFFull)        // global_addr[56:32]
          | 0x80000000u;                                // type=2 ("image")

  v8i g1;
  g1[0] = (int)0x00010000u;                             // wg_mask=0, data_size=1 (2B)
  g1[1] = (int)((td0 & 0xFFFFu) << 16);                 // tensor_dim0[15:0]
  g1[2] = (int)((td0 >> 16) | ((td1 & 0xFFFFu) << 16)); // tensor_dim0[31:16] | tensor_dim1[15:0]
  g1[3] = (int)((td1 >> 16) | ((td0 & 0xFFFFu) << 16)); // tensor_dim1[31:16] | tile_dim0
  g1[4] = (int)(td1 & 0xFFFFu);                         // tile_dim1 (tile_dim2=0)
  g1[5] = (int)(unsigned)stride;                        // tensor_dim0_stride[31:0]
  g1[6] = (int)((unsigned)(stride >> 32) & 0xFFFFu);    // tensor_dim0_stride[47:32]
  g1[7] = 0;

  v4i z4 = {0, 0, 0, 0};                                // groups 2/3 unused (2D)
  v8i z8 = {0, 0, 0, 0, 0, 0, 0, 0};
  __builtin_amdgcn_tensor_load_to_lds(g0, g1, z4, z4, z8, 0);
}

// ---------------------------------------------------------------------------
// fp32 -> bf16 elementwise convert
// ---------------------------------------------------------------------------
__global__ void cvt_kernel(const float* __restrict__ in, __bf16* __restrict__ out, int n) {
  int i = blockIdx.x * blockDim.x + threadIdx.x;
  if (i < n) out[i] = f2bf(in[i]);
}

// fp32 W[N][K] -> bf16 Wt[K][N]  (so GEMM B-panels are contiguous rows)
__global__ void wt_kernel(const float* __restrict__ w, __bf16* __restrict__ wt) {
  int i = blockIdx.x * blockDim.x + threadIdx.x;   // i = k*DM + n
  if (i >= DM * DM) return;
  int k = i / DM;
  int n = i - k * DM;
  wt[i] = f2bf(w[n * DM + k]);
}

// ---------------------------------------------------------------------------
// Y = X @ W^T + b.  Block (128 thr, 4 waves) computes a 64x64 tile:
// wave w owns rows [mb*64 + w*16, +16) x all 64 cols (4 accumulators).
// B panel (32 K x 64 N) staged in LDS by TDM, double-buffered.
// mode 0: V  -> out_bf [B,H,S,HD]
// mode 1: Q  -> RoPE -> out_bf [B,H,S,HD]
// mode 2: K  -> RoPE -> out_bf [B,H,HD,S]   (pre-transposed for attention)
// mode 3: out_f [MTOT,DM] fp32 (final projection)
// ---------------------------------------------------------------------------
__global__ void __launch_bounds__(128) proj_gemm(
    const __bf16* __restrict__ X,
    const __bf16* __restrict__ Wt,
    const float*  __restrict__ bias,
    __bf16* __restrict__ out_bf,
    float*  __restrict__ out_f,
    int mode)
{
  __shared__ __align__(32) __bf16 bsh[2][32 * 64];   // 8 KB double buffer

  const int lane  = threadIdx.x & 31;
  const int wid   = threadIdx.x >> 5;
  const int nb    = blockIdx.x & (DM / 64 - 1);     // 16 n-blocks
  const int mb    = blockIdx.x >> 4;                // 64 m-blocks
  const int l15   = lane & 15;
  const int hi    = lane >> 4;
  const int abase = hi ? 8 : 0;

  const __bf16* xrow = X + (size_t)(mb * 64 + wid * 16 + l15) * DM;
  const __bf16* wpan = Wt + (size_t)nb * 64;

  if (wid == 0)   // prologue: stage first B panel
    tdm_load_2d(bsh[0], wpan, 64, 32, DM);

  v8f acc[4] = {{}, {}, {}, {}};
  for (int k0 = 0, it = 0; k0 < DM; k0 += 32, ++it) {
    const int cur = it & 1;
    if (wid == 0) {
      if (k0 + 32 < DM) {
        tdm_load_2d(bsh[cur ^ 1], wpan + (size_t)(k0 + 32) * DM, 64, 32, DM);
        __builtin_amdgcn_s_wait_tensorcnt(1);   // current panel landed
      } else {
        __builtin_amdgcn_s_wait_tensorcnt(0);
      }
    }
    __syncthreads();

    // A fragment (16x32): row = lane&15, K split {0-7,16-23}/{8-15,24-31}
    v16bf a;
    {
      v8bf alo = *(const v8bf*)(xrow + k0 + abase);
      v8bf ahi = *(const v8bf*)(xrow + k0 + abase + 16);
#pragma unroll
      for (int e = 0; e < 8; ++e) { a[e] = alo[e]; a[e + 8] = ahi[e]; }
    }
    __builtin_prefetch(xrow + k0 + 256, 0, 0);

    const __bf16* bp = bsh[cur] + l15;   // B lane = N column of panel, K split like A
#pragma unroll
    for (int s = 0; s < 4; ++s) {
      v16bf b;
#pragma unroll
      for (int e = 0; e < 8; ++e) {
        b[e]     = bp[(abase + e) * 64 + s * 16];
        b[e + 8] = bp[(abase + 16 + e) * 64 + s * 16];
      }
      acc[s] = __builtin_amdgcn_wmma_f32_16x16x32_bf16(false, a, false, b,
                                                       (short)0, acc[s], false, false);
    }
    __syncthreads();   // panel fully consumed before TDM overwrites it
  }

  // ---- epilogue: bias (+ RoPE) + scatter, per 16-col sub-tile ----
#pragma unroll
  for (int s = 0; s < 4; ++s) {
    const int   n  = nb * 64 + s * 16 + l15;   // C/D lane = N
    const float bv = bias[n];
    float vals[8];
#pragma unroll
    for (int e = 0; e < 8; ++e) vals[e] = acc[s][e] + bv;

    if (mode == 3) {
#pragma unroll
      for (int e = 0; e < 8; ++e) {
        int m = mb * 64 + wid * 16 + e + hi * 8;
        out_f[(size_t)m * DM + n] = vals[e];
      }
      continue;
    }

    const int h = n >> 6;
    const int d = n & 63;
    float res[8];
    if (mode == 0) {
#pragma unroll
      for (int e = 0; e < 8; ++e) res[e] = vals[e];
    } else {
      // RoPE: inv_freq = theta^(-(d&~1)/HD); pair (even,odd) via shfl_xor(1)
      const float inv   = __expf(-(float)(d & ~1) * (9.210340371976184f / 64.0f));
      const int   deven = !(d & 1);
#pragma unroll
      for (int e = 0; e < 8; ++e) {
        int m = mb * 64 + wid * 16 + e + hi * 8;
        int srow = m & (SEQ - 1);
        float sn, cs;
        __sincosf((float)srow * inv, &sn, &cs);
        float self  = vals[e];
        float other = __shfl_xor(self, 1, 32);
        res[e] = deven ? (self * cs - other * sn)
                       : (other * sn + self * cs);
      }
    }
#pragma unroll
    for (int e = 0; e < 8; ++e) {
      int m = mb * 64 + wid * 16 + e + hi * 8;
      int srow = m & (SEQ - 1);
      int bidx = m >> 11;
      size_t head = (size_t)(bidx * NH + h);
      size_t idx = (mode == 2) ? (head * HD + d) * SEQ + srow
                               : (head * SEQ + srow) * HD + d;
      out_bf[idx] = f2bf(res[e]);
    }
  }
}

// ---------------------------------------------------------------------------
// Flash-style attention. Block = 4 waves over the same (b,h): waves share the
// K/V 32-key panels (TDM -> LDS, double-buffered); each wave owns 16 q-rows.
// Q: [B,H,S,HD], Kt: [B,H,HD,S], V: [B,H,S,HD], O: [B,S,DM] bf16.
// ---------------------------------------------------------------------------
__global__ void __launch_bounds__(128) attn_kernel(
    const __bf16* __restrict__ Q,
    const __bf16* __restrict__ Kt,
    const __bf16* __restrict__ V,
    __bf16* __restrict__ O)
{
  __shared__ __align__(32) __bf16 ksh[2][HD * 32];   // 8 KB: [d][32 keys]
  __shared__ __align__(32) __bf16 vsh[2][32 * HD];   // 8 KB: [key][d]
  __shared__ __align__(32) __bf16 psh[4][16 * 32];   // 4 KB per-wave P scratch

  const int lane  = threadIdx.x & 31;
  const int wid   = threadIdx.x >> 5;
  const int l15   = lane & 15;
  const int hi    = lane >> 4;
  const int abase = hi ? 8 : 0;

  const int bh = blockIdx.x >> 5;        // 0..31  (b*NH + h)
  const int qg = blockIdx.x & 31;
  const int mt = qg * 4 + wid;           // q tile, 0..127
  const int b  = bh >> 4;
  const int h  = bh & 15;

  const __bf16* q  = Q  + (size_t)bh * SEQ * HD;
  const __bf16* kt = Kt + (size_t)bh * HD * SEQ;
  const __bf16* vv = V  + (size_t)bh * SEQ * HD;
  __bf16* ps = psh[wid];

  // Preload Q A-fragments for K = 0..63 (two 16x32 fragments)
  v16bf aq[2];
  const __bf16* qrow = q + (size_t)(mt * 16 + l15) * HD;
#pragma unroll
  for (int t = 0; t < 2; ++t) {
    v8bf lo = *(const v8bf*)(qrow + t * 32 + abase);
    v8bf hh = *(const v8bf*)(qrow + t * 32 + abase + 16);
#pragma unroll
    for (int e = 0; e < 8; ++e) { aq[t][e] = lo[e]; aq[t][e + 8] = hh[e]; }
  }

  if (wid == 0) {   // prologue: stage first K/V panels
    tdm_load_2d(ksh[0], kt, 32, HD, SEQ);       // 64 d-rows x 32 keys
    tdm_load_2d(vsh[0], vv, HD, 32, HD);        // 32 key-rows x 64 d
  }

  float mrow[8], lrow[8];
  v8f o0 = {}, o1 = {}, o2 = {}, o3 = {};
#pragma unroll
  for (int e = 0; e < 8; ++e) { mrow[e] = -1e30f; lrow[e] = 0.0f; }

  for (int n0 = 0, it = 0; n0 < SEQ; n0 += 32, ++it) {
    const int cur = it & 1;
    if (wid == 0) {
      if (n0 + 32 < SEQ) {
        tdm_load_2d(ksh[cur ^ 1], kt + n0 + 32, 32, HD, SEQ);
        tdm_load_2d(vsh[cur ^ 1], vv + (size_t)(n0 + 32) * HD, HD, 32, HD);
        __builtin_amdgcn_s_wait_tensorcnt(2);   // current pair landed
      } else {
        __builtin_amdgcn_s_wait_tensorcnt(0);
      }
    }
    __syncthreads();

    // ---- scores: 16x32 = two C fragments; K=HD in two 32-steps ----
    v8f s0 = {}, s1 = {};
#pragma unroll
    for (int t = 0; t < 2; ++t) {
      const __bf16* kp = ksh[cur] + (t * 32) * 32 + l15;   // B lane = key column, d split like A
      v16bf bk0, bk1;
#pragma unroll
      for (int e = 0; e < 8; ++e) {
        bk0[e]     = kp[(abase + e) * 32];
        bk0[e + 8] = kp[(abase + 16 + e) * 32];
        bk1[e]     = kp[(abase + e) * 32 + 16];
        bk1[e + 8] = kp[(abase + 16 + e) * 32 + 16];
      }
      s0 = __builtin_amdgcn_wmma_f32_16x16x32_bf16(false, aq[t], false, bk0,
                                                   (short)0, s0, false, false);
      s1 = __builtin_amdgcn_wmma_f32_16x16x32_bf16(false, aq[t], false, bk1,
                                                   (short)0, s1, false, false);
    }

    // ---- online softmax (rows span 16 lanes; masks 1..8 stay in-group) ----
    float p0[8], p1[8], alpha[8];
#pragma unroll
    for (int e = 0; e < 8; ++e) {
      float a0 = s0[e] * 0.125f;   // 1/sqrt(HD)
      float a1 = s1[e] * 0.125f;
      float t_ = fmaxf(a0, a1);
#pragma unroll
      for (int msk = 1; msk < 16; msk <<= 1)
        t_ = fmaxf(t_, __shfl_xor(t_, msk, 32));
      float mn = fmaxf(mrow[e], t_);
      alpha[e] = __expf(mrow[e] - mn);
      mrow[e]  = mn;
      p0[e] = __expf(a0 - mn);
      p1[e] = __expf(a1 - mn);
      float rs = p0[e] + p1[e];
#pragma unroll
      for (int msk = 1; msk < 16; msk <<= 1)
        rs += __shfl_xor(rs, msk, 32);
      lrow[e] = lrow[e] * alpha[e] + rs;
    }
#pragma unroll
    for (int e = 0; e < 8; ++e) {
      o0[e] *= alpha[e]; o1[e] *= alpha[e];
      o2[e] *= alpha[e]; o3[e] *= alpha[e];
    }

    // ---- P: C layout -> A layout via per-wave LDS (same-wave DS in-order) ----
#pragma unroll
    for (int e = 0; e < 8; ++e) {
      int r = e + hi * 8;
      ps[r * 32 + l15]      = f2bf(p0[e]);
      ps[r * 32 + 16 + l15] = f2bf(p1[e]);
    }
    v16bf ap;
    {
      const __bf16* prow = ps + l15 * 32;
      v8bf lo = *(const v8bf*)(prow + abase);
      v8bf hh = *(const v8bf*)(prow + abase + 16);
#pragma unroll
      for (int e = 0; e < 8; ++e) { ap[e] = lo[e]; ap[e + 8] = hh[e]; }
    }

    // ---- o += P @ V : 4 d-tiles of 16, V from LDS ----
    const __bf16* vp = vsh[cur] + l15;   // B lane = d column, key split like A
#pragma unroll
    for (int dt = 0; dt < 4; ++dt) {
      v16bf bvf;
#pragma unroll
      for (int e = 0; e < 8; ++e) {
        bvf[e]     = vp[(abase + e) * HD + dt * 16];
        bvf[e + 8] = vp[(abase + 16 + e) * HD + dt * 16];
      }
      v8f acc = (dt == 0) ? o0 : (dt == 1) ? o1 : (dt == 2) ? o2 : o3;
      acc = __builtin_amdgcn_wmma_f32_16x16x32_bf16(false, ap, false, bvf,
                                                    (short)0, acc, false, false);
      if      (dt == 0) o0 = acc;
      else if (dt == 1) o1 = acc;
      else if (dt == 2) o2 = acc;
      else              o3 = acc;
    }
    __syncthreads();   // panels consumed before TDM overwrites them
  }

  // ---- normalize and store to [B,S,DM] ----
#pragma unroll
  for (int e = 0; e < 8; ++e) {
    int r = e + hi * 8;
    int srow = mt * 16 + r;
    float invl = 1.0f / lrow[e];
    size_t base = ((size_t)b * SEQ + srow) * DM + h * HD;
    O[base +  0 + l15] = f2bf(o0[e] * invl);
    O[base + 16 + l15] = f2bf(o1[e] * invl);
    O[base + 32 + l15] = f2bf(o2[e] * invl);
    O[base + 48 + l15] = f2bf(o3[e] * invl);
  }
}

// ---------------------------------------------------------------------------
// Host-side launch
// ---------------------------------------------------------------------------
extern "C" void kernel_launch(void* const* d_in, const int* in_sizes, int n_in,
                              void* d_out, int out_size, void* d_ws, size_t ws_size,
                              hipStream_t stream) {
  const float* q  = (const float*)d_in[0];
  const float* k  = (const float*)d_in[1];
  const float* v  = (const float*)d_in[2];
  const float* wq = (const float*)d_in[3];
  const float* bq = (const float*)d_in[4];
  const float* wk = (const float*)d_in[5];
  const float* bk = (const float*)d_in[6];
  const float* wv = (const float*)d_in[7];
  const float* bv = (const float*)d_in[8];
  const float* wo = (const float*)d_in[9];
  const float* bo = (const float*)d_in[10];
  float* out = (float*)d_out;

  const size_t XN = (size_t)MTOT * DM;   // 4,194,304 elements
  const size_t WN = (size_t)DM * DM;     // 1,048,576 elements

  __bf16* p   = (__bf16*)d_ws;
  __bf16* xq  = p; p += XN;
  __bf16* xk  = p; p += XN;
  __bf16* xv  = p; p += XN;
  __bf16* wqt = p; p += WN;
  __bf16* wkt = p; p += WN;
  __bf16* wvt = p; p += WN;
  __bf16* wot = p; p += WN;
  __bf16* qr  = p; p += XN;   // [B,H,S,HD]  Q after RoPE
  __bf16* ktb = p; p += XN;   // [B,H,HD,S]  K after RoPE, transposed
  __bf16* vpb = p; p += XN;   // [B,H,S,HD]
  __bf16* ao  = p; p += XN;   // [B,S,DM] attention output

  cvt_kernel<<<(int)((XN + 255) / 256), 256, 0, stream>>>(q, xq, (int)XN);
  cvt_kernel<<<(int)((XN + 255) / 256), 256, 0, stream>>>(k, xk, (int)XN);
  cvt_kernel<<<(int)((XN + 255) / 256), 256, 0, stream>>>(v, xv, (int)XN);
  wt_kernel<<<(int)((WN + 255) / 256), 256, 0, stream>>>(wq, wqt);
  wt_kernel<<<(int)((WN + 255) / 256), 256, 0, stream>>>(wk, wkt);
  wt_kernel<<<(int)((WN + 255) / 256), 256, 0, stream>>>(wv, wvt);
  wt_kernel<<<(int)((WN + 255) / 256), 256, 0, stream>>>(wo, wot);

  const int gemm_blocks = (MTOT / 64) * (DM / 64);   // 1024
  proj_gemm<<<gemm_blocks, 128, 0, stream>>>(xq, wqt, bq, qr,  nullptr, 1);
  proj_gemm<<<gemm_blocks, 128, 0, stream>>>(xk, wkt, bk, ktb, nullptr, 2);
  proj_gemm<<<gemm_blocks, 128, 0, stream>>>(xv, wvt, bv, vpb, nullptr, 0);

  attn_kernel<<<NBAT * NH * (SEQ / 16 / 4), 128, 0, stream>>>(qr, ktb, vpb, ao);

  proj_gemm<<<gemm_blocks, 128, 0, stream>>>(ao, wot, bo, nullptr, out, 3);
}
